// GNNEncoder_90924457657029
// MI455X (gfx1250) — matrix operand order repaired and checked by reading, not compiled, not run on hardware
//
#include <hip/hip_runtime.h>
#include <hip/hip_bf16.h>
#include <math.h>

#define N_NODES 20000
#define N_EDGES 640000
#define HEADS   4
#define DHEAD   64
#define CDIM    256          // HEADS * DHEAD
#define LAYERS  4
#define HIDDEN  512          // (LAYERS*CDIM)/2
#define G4      2048         // 4*HIDDEN
#define EPS_BN  1e-5f
#define NEG_SLOPE 0.2f

typedef __bf16 bf16;
typedef __attribute__((ext_vector_type(16))) __bf16 v16bf;
typedef __attribute__((ext_vector_type(8)))  __bf16 v8bf;
typedef __attribute__((ext_vector_type(8)))  float  v8f;

// ---------------------------------------------------------------------------
// WMMA bf16 GEMM:  Y[n,m] = X[n,K] @ Wt[m,K]^T  (+ bias[m]) (+= if accumulate)
// block = 256 threads (8 waves). Block tile 128x128, wave tile 32x64
// (2x4 of 16x16), K templated (256/512) and unrolled x2.
// Per wave per K-step: 12 ds_load_b128 feed 8 v_wmma_f32_16x16x32_bf16.
// ---------------------------------------------------------------------------
__device__ inline v8f wmma_bf16(v16bf a, v16bf b, v8f c) {
  return __builtin_amdgcn_wmma_f32_16x16x32_bf16(
      /*neg_a=*/false, a, /*neg_b=*/false, b,
      /*c_mod=*/(short)0, c, /*reuse_a=*/false, /*reuse_b=*/false);
}

template <int K>
__global__ __launch_bounds__(256) void gemm_bf16_wmma(
    const bf16* __restrict__ X, const bf16* __restrict__ Wt,
    float* __restrict__ Y, const float* __restrict__ bias,
    int n, int m, int accumulate) {
  // padded stride 40 bf16 = 80 bytes (multiple of 16 -> aligned b128 LDS ops)
  __shared__ bf16 lx[128][40];
  __shared__ bf16 lw[128][40];

  const int tid  = threadIdx.x;
  const int lane = tid & 31;
  const int wave = tid >> 5;
  const int wr   = wave >> 1;   // 0..3 row group (32 rows each)
  const int wc   = wave & 1;    // 0..1 col group (64 cols each)
  const int row0 = blockIdx.y * 128;
  const int col0 = blockIdx.x * 128;

  v8f acc[2][4] = {};

  // staging: each thread moves 16 bf16 of X and 16 bf16 of W per K-step
  const int srow = tid >> 1, sseg = tid & 1;      // 128 rows x 2 segs(16)
  const int xg   = row0 + srow;
  const bf16* Xrow = X + (size_t)xg * K + sseg * 16;
  const bf16* Wrow = Wt + (size_t)(col0 + srow) * K + sseg * 16;  // m % 128 == 0
  const bool xok = (xg < n);
  const int kh  = lane >> 4;    // K-half select per WMMA register layout
  const int l15 = lane & 15;

#pragma unroll 2
  for (int k0 = 0; k0 < K; k0 += 32) {
    v8bf x0 = {}, x1 = {};
    if (xok) {
      x0 = *(const v8bf*)(Xrow + k0);
      x1 = *(const v8bf*)(Xrow + k0 + 8);
      if (k0 + 32 < K) __builtin_prefetch(Xrow + k0 + 32, 0, 1);
    }
    v8bf w0 = *(const v8bf*)(Wrow + k0);
    v8bf w1 = *(const v8bf*)(Wrow + k0 + 8);
    if (k0 + 32 < K) __builtin_prefetch(Wrow + k0 + 32, 0, 1);

    *(v8bf*)&lx[srow][sseg * 16]     = x0;
    *(v8bf*)&lx[srow][sseg * 16 + 8] = x1;
    *(v8bf*)&lw[srow][sseg * 16]     = w0;
    *(v8bf*)&lw[srow][sseg * 16 + 8] = w1;
    __syncthreads();

    v16bf A[2], B[4];
#pragma unroll
    for (int ti = 0; ti < 2; ++ti) {
      // A 16x32 bf16 layout: lanes<16 hold K {0..7,16..23}, lanes>=16 {8..15,24..31}
      const bf16* p = &lx[wr * 32 + ti * 16 + l15][kh * 8];
      v8bf lo = *(const v8bf*)p;
      v8bf hi = *(const v8bf*)(p + 16);
      A[ti] = __builtin_shufflevector(lo, hi, 0, 1, 2, 3, 4, 5, 6, 7,
                                      8, 9, 10, 11, 12, 13, 14, 15);
    }
#pragma unroll
    for (int tj = 0; tj < 4; ++tj) {
      // B 32x16: lanes<16 -> N=lane, K {0..7,16..23}; lanes>=16 -> K {8..15,24..31}
      const bf16* p = &lw[wc * 64 + tj * 16 + l15][kh * 8];
      v8bf lo = *(const v8bf*)p;
      v8bf hi = *(const v8bf*)(p + 16);
      B[tj] = __builtin_shufflevector(lo, hi, 0, 1, 2, 3, 4, 5, 6, 7,
                                      8, 9, 10, 11, 12, 13, 14, 15);
    }
#pragma unroll
    for (int ti = 0; ti < 2; ++ti)
#pragma unroll
      for (int tj = 0; tj < 4; ++tj)
        acc[ti][tj] = wmma_bf16(A[ti], B[tj], acc[ti][tj]);
    __syncthreads();
  }

  // Epilogue. C/D layout: VGPR r -> M = r (lanes<16) or r+8 (lanes>=16), N = lane&15.
#pragma unroll
  for (int ti = 0; ti < 2; ++ti) {
#pragma unroll
    for (int tj = 0; tj < 4; ++tj) {
      const int col = col0 + wc * 64 + tj * 16 + l15;
#pragma unroll
      for (int r = 0; r < 8; ++r) {
        const int row = row0 + wr * 32 + ti * 16 + r + kh * 8;
        if (row < n) {
          float v = acc[ti][tj][r];
          if (bias) v += bias[col];
          size_t o = (size_t)row * m + col;
          if (accumulate) v += Y[o];
          Y[o] = v;
        }
      }
    }
  }
}

// ---------------------------------------------------------------------------
// Utility kernels
// ---------------------------------------------------------------------------
__global__ void fill_f32(float* p, float v, int count) {
  int i = blockIdx.x * blockDim.x + threadIdx.x;
  if (i < count) p[i] = v;
}
__global__ void fill_u16(unsigned short* p, unsigned short v, int count) {
  int i = blockIdx.x * blockDim.x + threadIdx.x;
  if (i < count) p[i] = v;
}
__global__ void to_bf16_kernel(const float* __restrict__ a, bf16* __restrict__ b, int count) {
  int i = blockIdx.x * blockDim.x + threadIdx.x;
  if (i < count) b[i] = (bf16)a[i];
}
// W: [nmat][K][M] f32  ->  Wt: [nmat][M][K] bf16
__global__ void transpose_to_bf16(const float* __restrict__ W, bf16* __restrict__ Wt,
                                  int nmat, int K, int M) {
  int i = blockIdx.x * blockDim.x + threadIdx.x;
  int total = nmat * K * M;
  if (i >= total) return;
  int l = i / (K * M), r = i % (K * M);
  int k = r / M, mm = r % M;
  Wt[(size_t)l * K * M + (size_t)mm * K + k] = (bf16)W[i];
}
__global__ void add_vec(const float* a, const float* b, float* o, int count) {
  int i = blockIdx.x * blockDim.x + threadIdx.x;
  if (i < count) o[i] = a[i] + b[i];
}

// ---------------------------------------------------------------------------
// GATv2 kernels
// ---------------------------------------------------------------------------
__device__ inline void atomicMaxF(float* addr, float val) {
  unsigned int* a = (unsigned int*)addr;
  unsigned int old = *a;
  while (val > __uint_as_float(old)) {
    unsigned int assumed = old;
    old = atomicCAS(a, assumed, __float_as_uint(val));
    if (old == assumed) break;
  }
}

// layer-0 input transform: x (N,2) @ W (2,C)
__global__ void transform0(const float* __restrict__ x,
                           const float* __restrict__ Wl0, const float* __restrict__ Wr0,
                           float* __restrict__ xl, float* __restrict__ xr) {
  int idx = blockIdx.x * blockDim.x + threadIdx.x;
  if (idx >= N_NODES * CDIM) return;
  int nn = idx / CDIM, c = idx % CDIM;
  float a = x[nn * 2 + 0], b = x[nn * 2 + 1];
  xl[idx] = a * Wl0[c] + b * Wl0[CDIM + c];
  xr[idx] = a * Wr0[c] + b * Wr0[CDIM + c];
}

__global__ __launch_bounds__(256) void edge_logits_k(
    const int* __restrict__ ei, const float* __restrict__ ea,
    const float* __restrict__ xl, const float* __restrict__ xr,
    const float* __restrict__ We, const float* __restrict__ att,
    float* __restrict__ logits, float* __restrict__ mx) {
  int idx = blockIdx.x * blockDim.x + threadIdx.x;
  if (idx >= N_EDGES * HEADS) return;
  int e = idx >> 2, h = idx & 3;
  int src = ei[e], dst = ei[N_EDGES + e];
  float eav = ea[e];
  const float4* pl = (const float4*)(xl + (size_t)src * CDIM + h * DHEAD);
  const float4* pr = (const float4*)(xr + (size_t)dst * CDIM + h * DHEAD);
  const float4* pw = (const float4*)(We + h * DHEAD);
  const float4* pa = (const float4*)(att + h * DHEAD);
  float s = 0.f;
#pragma unroll 4
  for (int q = 0; q < DHEAD / 4; ++q) {
    float4 a = pl[q], b = pr[q], w = pw[q], t = pa[q];
    float v0 = a.x + b.x + eav * w.x; v0 = v0 > 0.f ? v0 : NEG_SLOPE * v0;
    float v1 = a.y + b.y + eav * w.y; v1 = v1 > 0.f ? v1 : NEG_SLOPE * v1;
    float v2 = a.z + b.z + eav * w.z; v2 = v2 > 0.f ? v2 : NEG_SLOPE * v2;
    float v3 = a.w + b.w + eav * w.w; v3 = v3 > 0.f ? v3 : NEG_SLOPE * v3;
    s += v0 * t.x + v1 * t.y + v2 * t.z + v3 * t.w;
  }
  logits[idx] = s;
  atomicMaxF(&mx[dst * HEADS + h], s);
}

__global__ __launch_bounds__(256) void edge_softmax_k(
    const int* __restrict__ ei, float* __restrict__ logits,
    const float* __restrict__ mx, float* __restrict__ den) {
  int idx = blockIdx.x * blockDim.x + threadIdx.x;
  if (idx >= N_EDGES * HEADS) return;
  int e = idx >> 2, h = idx & 3;
  int dst = ei[N_EDGES + e];
  float exv = __expf(logits[idx] - mx[dst * HEADS + h]);
  logits[idx] = exv;   // reuse as numerator
  atomicAdd(&den[dst * HEADS + h], exv);
}

__global__ __launch_bounds__(256) void edge_aggregate_k(
    const int* __restrict__ ei, const float* __restrict__ ex,
    const float* __restrict__ den, const float* __restrict__ xl,
    float* __restrict__ agg) {
  int idx = blockIdx.x * blockDim.x + threadIdx.x;
  const int QC = CDIM / 4;
  if (idx >= N_EDGES * QC) return;
  int e = idx / QC, q = idx % QC;
  int c0 = q * 4, h = c0 >> 6;
  int src = ei[e], dst = ei[N_EDGES + e];
  float alpha = ex[e * HEADS + h] / den[dst * HEADS + h];
  float4 v = *(const float4*)(xl + (size_t)src * CDIM + c0);
  float* o = agg + (size_t)dst * CDIM + c0;
  atomicAdd(o + 0, alpha * v.x);
  atomicAdd(o + 1, alpha * v.y);
  atomicAdd(o + 2, alpha * v.z);
  atomicAdd(o + 3, alpha * v.w);
}

__global__ void finalize_layer(const float* __restrict__ agg, const float* __restrict__ b,
                               float* __restrict__ xs_l, bf16* __restrict__ xsbf_l) {
  int idx = blockIdx.x * blockDim.x + threadIdx.x;
  if (idx >= N_NODES * CDIM) return;
  float v = agg[idx] + b[idx % CDIM];
  v = v > 0.f ? v : 0.f;  // relu
  xs_l[idx] = v;
  xsbf_l[idx] = (bf16)v;
}

// ---------------------------------------------------------------------------
// LSTM + JK kernels
// ---------------------------------------------------------------------------
__device__ inline float sigmoidf(float x) { return 1.f / (1.f + __expf(-x)); }

__global__ void lstm_cell(const float* __restrict__ gates, float* __restrict__ cst,
                          float* __restrict__ hst, bf16* __restrict__ hbf) {
  int idx = blockIdx.x * blockDim.x + threadIdx.x;
  if (idx >= N_NODES * HIDDEN) return;
  int nn = idx / HIDDEN, j = idx % HIDDEN;
  const float* g = gates + (size_t)nn * G4;
  float ig = sigmoidf(g[j]);
  float fg = sigmoidf(g[HIDDEN + j]);
  float gg = tanhf(g[2 * HIDDEN + j]);
  float og = sigmoidf(g[3 * HIDDEN + j]);
  float c = fg * cst[idx] + ig * gg;
  float h = og * tanhf(c);
  cst[idx] = c;
  hst[idx] = h;
  hbf[idx] = (bf16)h;
}

// one wave per node: scores_l[n] += dot(h[n,:], jkw[:HIDDEN part])
__global__ __launch_bounds__(128) void jk_score(const float* __restrict__ hst,
                                                const float* __restrict__ jkw,
                                                float* __restrict__ scores_l) {
  int wave = threadIdx.x >> 5, lane = threadIdx.x & 31;
  int nn = blockIdx.x * 4 + wave;
  if (nn >= N_NODES) return;
  const float* hp = hst + (size_t)nn * HIDDEN;
  float s = 0.f;
  for (int j = lane; j < HIDDEN; j += 32) s += hp[j] * jkw[j];
#pragma unroll
  for (int off = 16; off > 0; off >>= 1) s += __shfl_xor(s, off, 32);
  if (lane == 0) scores_l[nn] += s;
}

__global__ void jk_combine(const float* __restrict__ scores, const float* __restrict__ xs,
                           float* __restrict__ comb) {
  int idx = blockIdx.x * blockDim.x + threadIdx.x;
  if (idx >= N_NODES * CDIM) return;
  int nn = idx / CDIM;
  const size_t NC = (size_t)N_NODES * CDIM;
  float s0 = scores[nn], s1 = scores[N_NODES + nn];
  float s2 = scores[2 * N_NODES + nn], s3 = scores[3 * N_NODES + nn];
  float m = fmaxf(fmaxf(s0, s1), fmaxf(s2, s3));
  float e0 = __expf(s0 - m), e1 = __expf(s1 - m), e2 = __expf(s2 - m), e3 = __expf(s3 - m);
  float inv = 1.f / (e0 + e1 + e2 + e3);
  comb[idx] = inv * (e0 * xs[idx] + e1 * xs[NC + idx] +
                     e2 * xs[2 * NC + idx] + e3 * xs[3 * NC + idx]);
}

__global__ __launch_bounds__(256) void col_meanvar(const float* __restrict__ comb,
                                                   float* __restrict__ mean,
                                                   float* __restrict__ var) {
  int c = blockIdx.x, tid = threadIdx.x;
  __shared__ float s1[256], s2[256];
  float a = 0.f, b = 0.f;
  for (int nn = tid; nn < N_NODES; nn += 256) {
    float v = comb[(size_t)nn * CDIM + c];
    a += v; b += v * v;
  }
  s1[tid] = a; s2[tid] = b;
  __syncthreads();
  for (int off = 128; off > 0; off >>= 1) {
    if (tid < off) { s1[tid] += s1[tid + off]; s2[tid] += s2[tid + off]; }
    __syncthreads();
  }
  if (tid == 0) {
    float mu = s1[0] / (float)N_NODES;
    mean[c] = mu;
    var[c] = s2[0] / (float)N_NODES - mu * mu;
  }
}

__global__ void bn_out(const float* __restrict__ comb, const float* __restrict__ mean,
                       const float* __restrict__ var, const float* __restrict__ gamma,
                       const float* __restrict__ beta, float* __restrict__ out) {
  int idx = blockIdx.x * blockDim.x + threadIdx.x;
  if (idx >= N_NODES * CDIM) return;
  int c = idx % CDIM;
  out[idx] = (comb[idx] - mean[c]) * rsqrtf(var[c] + EPS_BN) * gamma[c] + beta[c];
}

// ---------------------------------------------------------------------------
// Driver
// ---------------------------------------------------------------------------
static inline unsigned cdiv(long long total, int bs) {
  return (unsigned)((total + bs - 1) / bs);
}

extern "C" void kernel_launch(void* const* d_in, const int* in_sizes, int n_in,
                              void* d_out, int out_size, void* d_ws, size_t ws_size,
                              hipStream_t stream) {
  (void)in_sizes; (void)n_in; (void)out_size; (void)ws_size;
  const float* x     = (const float*)d_in[0];
  const int*   ei    = (const int*)d_in[1];
  const float* ea    = (const float*)d_in[2];
  const float* Wl0   = (const float*)d_in[3];
  const float* Wr0   = (const float*)d_in[4];
  const float* We0   = (const float*)d_in[5];
  const float* att0  = (const float*)d_in[6];
  const float* b0    = (const float*)d_in[7];
  const float* WlR   = (const float*)d_in[8];
  const float* WrR   = (const float*)d_in[9];
  const float* WeR   = (const float*)d_in[10];
  const float* attR  = (const float*)d_in[11];
  const float* bRp   = (const float*)d_in[12];
  const float* Wih_f = (const float*)d_in[13];
  const float* Whh_f = (const float*)d_in[14];
  const float* bih_f = (const float*)d_in[15];
  const float* bhh_f = (const float*)d_in[16];
  const float* Wih_b = (const float*)d_in[17];
  const float* Whh_b = (const float*)d_in[18];
  const float* bih_b = (const float*)d_in[19];
  const float* bhh_b = (const float*)d_in[20];
  const float* jkw   = (const float*)d_in[21];
  const float* gamma = (const float*)d_in[23];
  const float* beta  = (const float*)d_in[24];
  float* out = (float*)d_out;

  const size_t NC = (size_t)N_NODES * CDIM;
  char* wp = (char*)d_ws;
  auto alloc = [&](size_t bytes) -> void* {
    void* p = (void*)wp;
    wp += (bytes + 255) & ~(size_t)255;
    return p;
  };
  float* xl     = (float*)alloc(NC * 4);
  float* xr     = (float*)alloc(NC * 4);
  float* agg    = (float*)alloc(NC * 4);
  float* logits = (float*)alloc((size_t)N_EDGES * HEADS * 4);
  float* mx     = (float*)alloc((size_t)N_NODES * HEADS * 4);
  float* den    = (float*)alloc((size_t)N_NODES * HEADS * 4);
  float* xs     = (float*)alloc((size_t)LAYERS * NC * 4);
  bf16*  xsbf   = (bf16*)alloc((size_t)LAYERS * NC * 2);
  bf16*  WlTb   = (bf16*)alloc((size_t)(LAYERS - 1) * CDIM * CDIM * 2);
  bf16*  WrTb   = (bf16*)alloc((size_t)(LAYERS - 1) * CDIM * CDIM * 2);
  bf16*  Wihbf  = (bf16*)alloc((size_t)G4 * CDIM * 2);
  bf16*  Whhbf  = (bf16*)alloc((size_t)G4 * HIDDEN * 2);
  bf16*  Wihbb  = (bf16*)alloc((size_t)G4 * CDIM * 2);
  bf16*  Whhbb  = (bf16*)alloc((size_t)G4 * HIDDEN * 2);
  float* bsum_f = (float*)alloc(G4 * 4);
  float* bsum_b = (float*)alloc(G4 * 4);
  float* gates  = (float*)alloc((size_t)N_NODES * G4 * 4);
  float* hst    = (float*)alloc((size_t)N_NODES * HIDDEN * 4);
  float* cst    = (float*)alloc((size_t)N_NODES * HIDDEN * 4);
  bf16*  hbf    = (bf16*)alloc((size_t)N_NODES * HIDDEN * 2);
  float* scores = (float*)alloc((size_t)LAYERS * N_NODES * 4);
  float* comb   = (float*)alloc(NC * 4);
  float* meanv  = (float*)alloc(CDIM * 4);
  float* varv   = (float*)alloc(CDIM * 4);

  auto gemm = [&](const bf16* X, const bf16* Wt, float* Y, const float* bias,
                  int n, int K, int m, int accum) {
    dim3 g(m / 128, (n + 127) / 128);
    if (K == 256)
      gemm_bf16_wmma<256><<<g, 256, 0, stream>>>(X, Wt, Y, bias, n, m, accum);
    else
      gemm_bf16_wmma<512><<<g, 256, 0, stream>>>(X, Wt, Y, bias, n, m, accum);
  };

  // --- weight prep (bf16 + transpose where needed) ---
  transpose_to_bf16<<<cdiv((long long)(LAYERS - 1) * CDIM * CDIM, 256), 256, 0, stream>>>(
      WlR, WlTb, LAYERS - 1, CDIM, CDIM);
  transpose_to_bf16<<<cdiv((long long)(LAYERS - 1) * CDIM * CDIM, 256), 256, 0, stream>>>(
      WrR, WrTb, LAYERS - 1, CDIM, CDIM);
  to_bf16_kernel<<<cdiv(G4 * CDIM, 256), 256, 0, stream>>>(Wih_f, Wihbf, G4 * CDIM);
  to_bf16_kernel<<<cdiv(G4 * HIDDEN, 256), 256, 0, stream>>>(Whh_f, Whhbf, G4 * HIDDEN);
  to_bf16_kernel<<<cdiv(G4 * CDIM, 256), 256, 0, stream>>>(Wih_b, Wihbb, G4 * CDIM);
  to_bf16_kernel<<<cdiv(G4 * HIDDEN, 256), 256, 0, stream>>>(Whh_b, Whhbb, G4 * HIDDEN);
  add_vec<<<cdiv(G4, 256), 256, 0, stream>>>(bih_f, bhh_f, bsum_f, G4);
  add_vec<<<cdiv(G4, 256), 256, 0, stream>>>(bih_b, bhh_b, bsum_b, G4);

  // --- GATv2 layers ---
  for (int l = 0; l < LAYERS; ++l) {
    if (l == 0) {
      transform0<<<cdiv(NC, 256), 256, 0, stream>>>(x, Wl0, Wr0, xl, xr);
    } else {
      gemm(xsbf + (size_t)(l - 1) * NC, WlTb + (size_t)(l - 1) * CDIM * CDIM,
           xl, nullptr, N_NODES, CDIM, CDIM, 0);
      gemm(xsbf + (size_t)(l - 1) * NC, WrTb + (size_t)(l - 1) * CDIM * CDIM,
           xr, nullptr, N_NODES, CDIM, CDIM, 0);
    }
    fill_f32<<<cdiv(N_NODES * HEADS, 256), 256, 0, stream>>>(mx, -3.0e38f, N_NODES * HEADS);
    fill_f32<<<cdiv(N_NODES * HEADS, 256), 256, 0, stream>>>(den, 0.f, N_NODES * HEADS);
    fill_f32<<<cdiv(NC, 256), 256, 0, stream>>>(agg, 0.f, (int)NC);

    const float* We  = l ? WeR + (size_t)(l - 1) * CDIM : We0;
    const float* att = l ? attR + (size_t)(l - 1) * HEADS * DHEAD : att0;
    const float* bb  = l ? bRp + (size_t)(l - 1) * CDIM : b0;

    edge_logits_k<<<cdiv((long long)N_EDGES * HEADS, 256), 256, 0, stream>>>(
        ei, ea, xl, xr, We, att, logits, mx);
    edge_softmax_k<<<cdiv((long long)N_EDGES * HEADS, 256), 256, 0, stream>>>(
        ei, logits, mx, den);
    edge_aggregate_k<<<cdiv((long long)N_EDGES * (CDIM / 4), 256), 256, 0, stream>>>(
        ei, logits, den, xl, agg);
    finalize_layer<<<cdiv(NC, 256), 256, 0, stream>>>(
        agg, bb, xs + (size_t)l * NC, xsbf + (size_t)l * NC);
  }

  // --- bidirectional JK-LSTM (scores only; hidden states not stored per layer) ---
  fill_f32<<<cdiv(LAYERS * N_NODES, 256), 256, 0, stream>>>(scores, 0.f, LAYERS * N_NODES);
  for (int dir = 0; dir < 2; ++dir) {
    const bf16* Wih = dir ? Wihbb : Wihbf;
    const bf16* Whh = dir ? Whhbb : Whhbf;
    const float* bs = dir ? bsum_b : bsum_f;
    const float* jp = jkw + (dir ? HIDDEN : 0);
    fill_f32<<<cdiv(N_NODES * HIDDEN, 256), 256, 0, stream>>>(hst, 0.f, N_NODES * HIDDEN);
    fill_f32<<<cdiv(N_NODES * HIDDEN, 256), 256, 0, stream>>>(cst, 0.f, N_NODES * HIDDEN);
    fill_u16<<<cdiv(N_NODES * HIDDEN, 256), 256, 0, stream>>>(
        (unsigned short*)hbf, (unsigned short)0, N_NODES * HIDDEN);
    for (int s = 0; s < LAYERS; ++s) {
      int t = dir ? (LAYERS - 1 - s) : s;
      gemm(xsbf + (size_t)t * NC, Wih, gates, bs, N_NODES, CDIM, G4, 0);
      gemm(hbf, Whh, gates, nullptr, N_NODES, HIDDEN, G4, 1);
      lstm_cell<<<cdiv(N_NODES * HIDDEN, 256), 256, 0, stream>>>(gates, cst, hst, hbf);
      jk_score<<<(N_NODES + 3) / 4, 128, 0, stream>>>(hst, jp, scores + (size_t)t * N_NODES);
    }
  }

  // --- JK combine + batch-norm ---
  jk_combine<<<cdiv(NC, 256), 256, 0, stream>>>(scores, xs, comb);
  col_meanvar<<<CDIM, 256, 0, stream>>>(comb, meanv, varv);
  bn_out<<<cdiv(NC, 256), 256, 0, stream>>>(comb, meanv, varv, gamma, beta, out);
}
